// NaiveModel_81243601371273
// MI455X (gfx1250) — matrix operand order repaired and checked by reading, not compiled, not run on hardware
//
#include <hip/hip_runtime.h>
#include <stdint.h>

#define B_    128
#define T_    512
#define FIN_  64
#define H_    1024
#define G4_   4096
#define NOUT_ 10

#define KT_   32                 // recurrent k-tiles (K=1024 / 32)
#define CKT_  4                  // k-tiles per staged LDS chunk
#define NC_   (KT_ / CKT_)       // 8 chunks
#define CH_SH (CKT_ * 4 * 512)   // shorts per chunk buffer (4 gates) = 8192 (16KB)
#define XP_SH (2 * 4 * 512)      // x-panel shorts (KT=2, 4 gates)     = 4096 (8KB)

typedef __attribute__((ext_vector_type(16))) __bf16       v16bf;
typedef __attribute__((ext_vector_type(8)))  float        v8f;
typedef __attribute__((ext_vector_type(4)))  unsigned int u32x4;
typedef __attribute__((ext_vector_type(4)))  int          i32x4;
typedef __attribute__((ext_vector_type(8)))  int          i32x8;

union Frag { v16bf v; u32x4 q[2]; };

__device__ __forceinline__ unsigned short f2bf(float f) {
  unsigned int u = __float_as_uint(f);
  u += 0x7FFFu + ((u >> 16) & 1u);      // round-to-nearest-even
  return (unsigned short)(u >> 16);
}
__device__ __forceinline__ float bf2f(unsigned short s) {
  return __uint_as_float(((unsigned int)s) << 16);
}
__device__ __forceinline__ float sigf(float x) {
  return 1.0f / (1.0f + __expf(-x));
}

#if __has_builtin(__builtin_amdgcn_tensor_load_to_lds)
#define HAVE_TDM 1
#else
#define HAVE_TDM 0
#endif

#if HAVE_TDM
// 2D TDM copy: nrows rows of row_dwords DWORDs, global row stride row_stride_dw
// DWORDs, into contiguous LDS at byte offset lds_off. D# per ISA ch.8.
__device__ __forceinline__ void tdm_load_2d(unsigned lds_off, const void* gsrc,
                                            unsigned row_dwords, unsigned nrows,
                                            unsigned row_stride_dw) {
  unsigned long long ga = (unsigned long long)(uintptr_t)gsrc;
  u32x4 g0;
  g0[0] = 1u;                                           // count=1 (valid user D#)
  g0[1] = lds_off;                                      // lds_addr (bytes)
  g0[2] = (unsigned)ga;                                 // global_addr[31:0]
  g0[3] = (unsigned)((ga >> 32) & 0x01FFFFFFu) | 0x80000000u;  // addr[56:32] | type=2
  i32x8 g1;
  g1[0] = (int)(2u << 16);                              // data_size = 4 bytes
  g1[1] = (int)((row_dwords & 0xFFFFu) << 16);          // tensor_dim0[15:0]
  g1[2] = (int)((row_dwords >> 16) | ((nrows & 0xFFFFu) << 16)); // dim0 hi | dim1 lo
  g1[3] = (int)((row_dwords & 0xFFFFu) << 16);          // tile_dim0
  g1[4] = (int)(nrows & 0xFFFFu);                       // tile_dim1 (tile_dim2=0)
  g1[5] = (int)row_stride_dw;                           // tensor_dim0_stride[31:0]
  g1[6] = 0;
  g1[7] = 0;
  i32x4 gz; gz[0] = 0; gz[1] = 0; gz[2] = 0; gz[3] = 0;
#if defined(__clang_major__) && (__clang_major__ >= 23)
  i32x8 gz8; for (int i = 0; i < 8; ++i) gz8[i] = 0;
  __builtin_amdgcn_tensor_load_to_lds(g0, g1, gz, gz, gz8, 0);
#else
  __builtin_amdgcn_tensor_load_to_lds(g0, g1, gz, gz, 0);
#endif
}
#endif

// ---------------- prep kernels ----------------

__global__ __launch_bounds__(256) void cvt_kernel(unsigned short* __restrict__ dst,
                                                  const float* __restrict__ src, int n) {
  int i = blockIdx.x * 256 + threadIdx.x;
  if (i < n) dst[i] = f2bf(src[i]);
}

// Wfused_f32[f, n] = sum_k dense_W[f,k] * Wk[k,n]   ([64,4096])
__global__ __launch_bounds__(256) void fuse_kernel(float* __restrict__ dst,
                                                   const float* __restrict__ dW,
                                                   const float* __restrict__ Wk) {
  int idx = blockIdx.x * 256 + threadIdx.x;
  if (idx >= FIN_ * G4_) return;
  int f = idx / G4_, n = idx % G4_;
  float s = 0.0f;
  for (int k = 0; k < H_; ++k) s += dW[f * H_ + k] * Wk[(size_t)k * G4_ + n];
  dst[idx] = s;
}

// bias_w[n] = b[n] + sum_k dense_b[k] * Wk[k,n]
__global__ __launch_bounds__(256) void bias_kernel(float* __restrict__ dst,
                                                   const float* __restrict__ b,
                                                   const float* __restrict__ db,
                                                   const float* __restrict__ Wk) {
  int n = blockIdx.x * 256 + threadIdx.x;
  if (n >= G4_) return;
  float s = b[n];
  for (int k = 0; k < H_; ++k) s += db[k] * Wk[(size_t)k * G4_ + n];
  dst[n] = s;
}

// Pack a [K, 4096] f32 weight (optionally A+B) into bf16 WMMA B-fragment order.
// dst index = ((nt*KT + kt)*32 + lane)*16 + j
// lane<16 : col = nt*16+lane,    K = kt*32 + j        (j=0..15)
// lane>=16: col = nt*16+lane-16, K = kt*32 + 16 + j
__global__ __launch_bounds__(256) void pack_kernel(unsigned short* __restrict__ dst,
                                                   const float* __restrict__ a,
                                                   const float* __restrict__ bsrc,
                                                   int KT) {
  int idx = blockIdx.x * 256 + threadIdx.x;
  int total = KT * 32 * G4_;
  if (idx >= total) return;
  int j    = idx & 15;
  int lane = (idx >> 4) & 31;
  int kt   = (idx >> 9) % KT;
  int nt   = (idx >> 9) / KT;
  int k = kt * 32 + (lane >> 4) * 16 + j;
  int n = nt * 16 + (lane & 15);
  float v = a[(size_t)k * G4_ + n];
  if (bsrc) v += bsrc[(size_t)k * G4_ + n];
  dst[idx] = f2bf(v);
}

__global__ __launch_bounds__(256) void zero_kernel(unsigned short* __restrict__ h,
                                                   float* __restrict__ c) {
  int i = blockIdx.x * 256 + threadIdx.x;
  if (i >= B_ * H_) return;
  h[i] = 0;
  c[i] = 0.0f;
}

// ---------------- recurrent step kernel ----------------
// Block ct = blockIdx.x owns 16 hidden columns; wave rt owns 16 batch rows.
// Weight panels (4 gates) for this block are streamed into LDS with the TDM,
// double-buffered in 16KB chunks, overlapped with WMMA on the previous chunk.
__global__ __launch_bounds__(256) void lstm_step(
    const unsigned short* __restrict__ hin,   // [128,1024] bf16
    unsigned short* __restrict__ hout,        // [128,1024] bf16
    float* __restrict__ c,                    // [128,1024] f32 (in/out)
    const unsigned short* __restrict__ whp,   // packed recurrent weights, KT=32
    const unsigned short* __restrict__ xin,   // optional input rows (bf16), ld=ldx
    const unsigned short* __restrict__ wxp,   // packed input weights, KT=2
    int ldx,
    const float* __restrict__ bias)           // [4096] gate-major i,f,g,o
{
  __shared__ __align__(16) unsigned short smem[2 * CH_SH + XP_SH];  // 40KB

  const int lane = threadIdx.x & 31;
  const int wave = threadIdx.x >> 5;
  const int ct   = blockIdx.x;                // hidden col tile 0..63
  const int r0   = wave << 4;                 // batch row base
  const int n0   = ct << 4;
  const int half = lane >> 4;
  const int l15  = lane & 15;

#if HAVE_TDM
  const unsigned lds0 = (unsigned)(uintptr_t)&smem[0];
  if (wave == 0) {
    if (xin)   // x-panel: 4 gate rows of 2KB, gate stride 64*2*512 shorts
      tdm_load_2d(lds0 + 2u * CH_SH * 2u, wxp + (size_t)ct * (2 * 512),
                  2 * 256, 4, 64 * 2 * 256);
    // chunk 0 of recurrent panel: 4 gate rows of CKT_*1KB, gate stride 64*KT_*512
    tdm_load_2d(lds0, whp + (size_t)ct * (KT_ * 512),
                CKT_ * 256, 4, 64 * KT_ * 256);
  }
#else
  if (xin) {
    for (int i = threadIdx.x; i < XP_SH / 8; i += 256) {
      int s = i * 8, g = s / 1024, off = s % 1024;
      ((u32x4*)&smem[2 * CH_SH])[i] =
          *(const u32x4*)&wxp[(size_t)(g * 64 + ct) * 1024 + off];
    }
  }
#endif

  v8f acc[4];
#pragma unroll
  for (int g = 0; g < 4; ++g)
#pragma unroll
    for (int j = 0; j < 8; ++j) acc[g][j] = 0.0f;

  // A fragment row base (contiguous k order used here, same as B pack: lanes 0-15 K{0..15},
  // lanes 16-31 K{16..31})
  const unsigned short* arow = hin + (size_t)(r0 + l15) * H_ + half * 16;

  for (int ch = 0; ch < NC_; ++ch) {
#if HAVE_TDM
    if (wave == 0) {
      if (ch + 1 < NC_) {
        tdm_load_2d(lds0 + (unsigned)(((ch + 1) & 1) * CH_SH * 2),
                    whp + ((size_t)ct * KT_ + (size_t)(ch + 1) * CKT_) * 512,
                    CKT_ * 256, 4, 64 * KT_ * 256);
        __builtin_amdgcn_s_wait_tensorcnt(1);   // chunk ch (and x-panel) resident
      } else {
        __builtin_amdgcn_s_wait_tensorcnt(0);
      }
    }
#else
    for (int i = threadIdx.x; i < CH_SH / 8; i += 256) {
      int s = i * 8, g = s / (CKT_ * 512), off = s % (CKT_ * 512);
      ((u32x4*)&smem[(ch & 1) * CH_SH])[i] =
          *(const u32x4*)&whp[((size_t)(g * 64 + ct) * KT_ + (size_t)ch * CKT_) * 512 + off];
    }
#endif
    __syncthreads();                            // publish chunk ch

    const unsigned short* buf = &smem[(ch & 1) * CH_SH];
#pragma unroll
    for (int kk = 0; kk < CKT_; ++kk) {
      const int kt = ch * CKT_ + kk;
      Frag a;
      a.q[0] = *(const u32x4*)(arow + kt * 32);
      a.q[1] = *(const u32x4*)(arow + kt * 32 + 8);
#pragma unroll
      for (int g = 0; g < 4; ++g) {
        const u32x4* bp = (const u32x4*)&buf[((g * CKT_ + kk) * 32 + lane) * 16];
        Frag bf;
        bf.q[0] = bp[0];
        bf.q[1] = bp[1];
        acc[g] = __builtin_amdgcn_wmma_f32_16x16x32_bf16(
            false, a.v, false, bf.v, (short)0, acc[g], false, false);
      }
    }
    __syncthreads();                            // all reads of buf done before reuse
  }

  // Input term (warmup only, K=64): x-panel already resident in LDS
  if (xin) {
    const unsigned short* xrow = xin + (size_t)(r0 + l15) * ldx + half * 16;
    const unsigned short* xb = &smem[2 * CH_SH];
#pragma unroll
    for (int kt = 0; kt < 2; ++kt) {
      Frag a;
      a.q[0] = *(const u32x4*)(xrow + kt * 32);
      a.q[1] = *(const u32x4*)(xrow + kt * 32 + 8);
#pragma unroll
      for (int g = 0; g < 4; ++g) {
        const u32x4* bp = (const u32x4*)&xb[((g * 2 + kt) * 32 + lane) * 16];
        Frag bf;
        bf.q[0] = bp[0];
        bf.q[1] = bp[1];
        acc[g] = __builtin_amdgcn_wmma_f32_16x16x32_bf16(
            false, a.v, false, bf.v, (short)0, acc[g], false, false);
      }
    }
  }

  // Pointwise LSTM update. C/D layout: VGPR j -> M = j + half*8, N = lane&15.
  const int n  = n0 + l15;
  const int mb = r0 + half * 8;
  const float bi  = bias[n];
  const float bff = bias[H_ + n];
  const float bg  = bias[2 * H_ + n];
  const float bo  = bias[3 * H_ + n];
#pragma unroll
  for (int j = 0; j < 8; ++j) {
    const int m = mb + j;
    const float zi = acc[0][j] + bi;
    const float zf = acc[1][j] + bff;
    const float zg = acc[2][j] + bg;
    const float zo = acc[3][j] + bo;
    const float cold = c[(size_t)m * H_ + n];
    const float cn = sigf(zf) * cold + sigf(zi) * tanhf(zg);
    c[(size_t)m * H_ + n] = cn;
    hout[(size_t)m * H_ + n] = f2bf(sigf(zo) * tanhf(cn));
  }
}

// ---------------- output projection ----------------
__global__ __launch_bounds__(256) void proj_kernel(const unsigned short* __restrict__ h,
                                                   const float* __restrict__ outW,
                                                   const float* __restrict__ outb,
                                                   float* __restrict__ out,
                                                   int tout, int OW) {
  int id = blockIdx.x * 256 + threadIdx.x;
  if (id >= B_ * NOUT_) return;
  int m = id / NOUT_, o = id % NOUT_;
  float s = outb[o];
  for (int k = 0; k < H_; ++k) s += bf2f(h[(size_t)m * H_ + k]) * outW[k * NOUT_ + o];
  out[((size_t)m * OW + tout) * NOUT_ + o] = s;
}

// ---------------- host launcher ----------------
extern "C" void kernel_launch(void* const* d_in, const int* in_sizes, int n_in,
                              void* d_out, int out_size, void* d_ws, size_t ws_size,
                              hipStream_t stream) {
  const float* x  = (const float*)d_in[0];
  const float* dW = (const float*)d_in[1];
  const float* db = (const float*)d_in[2];
  const float* Wk = (const float*)d_in[3];
  const float* Wr = (const float*)d_in[4];
  const float* b  = (const float*)d_in[5];
  const float* oW = (const float*)d_in[6];
  const float* ob = (const float*)d_in[7];
  float* out = (float*)d_out;
  const int OW = out_size / (B_ * NOUT_);   // = output_width

  char* p = (char*)d_ws;
  auto alloc = [&](size_t bytes) -> char* {
    char* r = p;
    p += (bytes + 255) & ~(size_t)255;
    return r;
  };
  unsigned short* xb   = (unsigned short*)alloc((size_t)B_ * T_ * FIN_ * 2);
  unsigned short* wrp  = (unsigned short*)alloc((size_t)H_ * G4_ * 2);
  unsigned short* wsp  = (unsigned short*)alloc((size_t)H_ * G4_ * 2);
  unsigned short* wxp  = (unsigned short*)alloc((size_t)FIN_ * G4_ * 2);
  float*          wf32 = (float*)alloc((size_t)FIN_ * G4_ * 4);
  float*          bw   = (float*)alloc((size_t)G4_ * 4);
  unsigned short* h0   = (unsigned short*)alloc((size_t)B_ * H_ * 2);
  unsigned short* h1   = (unsigned short*)alloc((size_t)B_ * H_ * 2);
  float*          c    = (float*)alloc((size_t)B_ * H_ * 4);

  cvt_kernel<<<(B_ * T_ * FIN_ + 255) / 256, 256, 0, stream>>>(xb, x, B_ * T_ * FIN_);
  fuse_kernel<<<(FIN_ * G4_ + 255) / 256, 256, 0, stream>>>(wf32, dW, Wk);
  pack_kernel<<<(H_ * G4_ + 255) / 256, 256, 0, stream>>>(wrp, Wr, nullptr, 32);
  pack_kernel<<<(H_ * G4_ + 255) / 256, 256, 0, stream>>>(wsp, Wk, Wr, 32);
  pack_kernel<<<(FIN_ * G4_ + 255) / 256, 256, 0, stream>>>(wxp, wf32, nullptr, 2);
  bias_kernel<<<(G4_ + 255) / 256, 256, 0, stream>>>(bw, b, db, Wk);
  zero_kernel<<<(B_ * H_ + 255) / 256, 256, 0, stream>>>(h0, c);

  // Warmup scan: 512 sequential steps; ping-pong h buffers for step isolation
  for (int t = 0; t < T_; ++t) {
    const unsigned short* hin = (t & 1) ? h1 : h0;
    unsigned short* hout      = (t & 1) ? h0 : h1;
    lstm_step<<<64, 256, 0, stream>>>(hin, hout, c, wrp,
                                      xb + (size_t)t * FIN_, wxp, T_ * FIN_, bw);
  }
  // after t=511 (odd), final h is in h0
  proj_kernel<<<(B_ * NOUT_ + 255) / 256, 256, 0, stream>>>(h0, oW, ob, out, 0, OW);

  // Autoregressive: input == h, so z = h @ (Wk+Wr) + b
  for (int s = 1; s < OW; ++s) {
    const unsigned short* hin = (s & 1) ? h0 : h1;
    unsigned short* hout      = (s & 1) ? h1 : h0;
    lstm_step<<<64, 256, 0, stream>>>(hin, hout, c, wsp, nullptr, nullptr, 0, b);
    proj_kernel<<<(B_ * NOUT_ + 255) / 256, 256, 0, stream>>>(hout, oW, ob, out, s, OW);
  }
}
